// TransformerConvNet_17334488006899
// MI455X (gfx1250) — matrix operand order repaired and checked by reading, not compiled, not run on hardware
//
#include <hip/hip_runtime.h>

typedef _Float16 f16;
typedef __attribute__((ext_vector_type(16))) _Float16 v16h;
typedef __attribute__((ext_vector_type(8)))  float    v8f;

// ---------------------------------------------------------------------------
// WMMA core: v_wmma_f32_16x16x32_f16, D = A(16x32) * B(32x16) + C
// ---------------------------------------------------------------------------
__device__ __forceinline__ v8f wmma_acc(v16h a, v16h b, v8f c) {
  return __builtin_amdgcn_wmma_f32_16x16x32_f16(
      false, a, false, b, (short)0, c, false, false);
}

// Fragment-major storage: one (k-step, n-tile) B tile = 512 f16, laid out
// [lane][i] so a lane's fragment is 32 contiguous bytes (2 x b128 loads).
__device__ __forceinline__ v16h load_b_packed(const f16* base, int ntn, int kt, int nt) {
  int lane = threadIdx.x & 31;
  return *(const v16h*)(base + (((size_t)kt * ntn + nt) * 32 + lane) * 16);
}
// Same layout for A tiles staged in LDS.
__device__ __forceinline__ v16h load_a_packed(const f16* base, int nkt, int rt, int kt) {
  int lane = threadIdx.x & 31;
  return *(const v16h*)(base + (((size_t)rt * nkt + kt) * 32 + lane) * 16);
}

// Inverse of the ISA 16-bit A-fragment layout: element (m, kk) of a 16x32 tile
// lives at lane*16+i. lanes 0-15: K {0..7,16..23}; lanes 16-31: K {8..15,24..31}.
__device__ __forceinline__ int afrag_idx(int m, int kk) {
  int half = ((kk & 15) >= 8) ? 1 : 0;
  int lane = m + 16 * half;
  int i = (kk < 16) ? (kk - 8 * half) : (8 + (kk - 16) - 8 * half);
  return lane * 16 + i;
}

// Orderable-uint encoding for float atomicMax.
__device__ __forceinline__ unsigned fkey(float x) {
  unsigned u = __float_as_uint(x);
  return (u & 0x80000000u) ? ~u : (u | 0x80000000u);
}
__device__ __forceinline__ float funkey(unsigned k) {
  unsigned u = (k & 0x80000000u) ? (k & 0x7fffffffu) : ~k;
  return __uint_as_float(u);
}

// ---------------------------------------------------------------------------
// Repack row-major KxN f32 weight -> fragment-major f16 tiles.
// out[((kt*(N/16)+nt)*32 + lane)*16 + i] = w[(kt*32 + (i&8)*2 + (lane>>4)*8 + (i&7))*N + nt*16 + (lane&15)]
// ---------------------------------------------------------------------------
__global__ void __launch_bounds__(256) pack_b_kernel(const float* __restrict__ w,
                                                     f16* __restrict__ out, int K, int Nn) {
  int gid = blockIdx.x * 256 + threadIdx.x;
  if (gid >= K * Nn) return;
  int i    = gid & 15;
  int lane = (gid >> 4) & 31;
  int tile = gid >> 9;
  int ntn  = Nn >> 4;
  int nt = tile % ntn, kt = tile / ntn;
  int Kidx = kt * 32 + (i & 8) * 2 + (lane >> 4) * 8 + (i & 7);
  int Nidx = nt * 16 + (lane & 15);
  out[gid] = (f16)w[(size_t)Kidx * Nn + Nidx];
}

// ---------------------------------------------------------------------------
// LayerNorm over 64 rows x 128 cols in LDS. 256 threads = 64 rows x 4 threads.
// red: 640 floats. Optionally writes fragment-major f16 copy for next GEMM.
// ---------------------------------------------------------------------------
__device__ __forceinline__ void ln_rows64x128(float* buf, const float* g, const float* b,
                                              float* red, f16* outP) {
  int tid = threadIdx.x;
  int row = tid >> 2;
  int sub = tid & 3;
  float s = 0.f, ss = 0.f;
#pragma unroll
  for (int i = 0; i < 32; ++i) {
    float v = buf[row * 128 + sub * 32 + i];
    s += v; ss += v * v;
  }
  red[tid] = s;
  red[256 + tid] = ss;
  __syncthreads();
  if (sub == 0) {
    float S  = red[tid] + red[tid + 1] + red[tid + 2] + red[tid + 3];
    float SS = red[256 + tid] + red[257 + tid] + red[258 + tid] + red[259 + tid];
    float mu  = S * (1.f / 128.f);
    float var = SS * (1.f / 128.f) - mu * mu;
    red[512 + row] = mu;
    red[576 + row] = rsqrtf(var + 1e-5f);
  }
  __syncthreads();
  float mu = red[512 + row], r = red[576 + row];
  int rt = row >> 4, m = row & 15;
#pragma unroll
  for (int i = 0; i < 32; ++i) {
    int c = sub * 32 + i;
    float v = (buf[row * 128 + c] - mu) * r * g[c] + b[c];
    buf[row * 128 + c] = v;
    if (outP) {
      int kt = c >> 5, kk = c & 31;
      outP[(rt * 4 + kt) * 512 + afrag_idx(m, kk)] = (f16)v;
    }
  }
  __syncthreads();
}

// ---------------------------------------------------------------------------
// Embed: per-edge 2-layer MLP, scatter-sum to dst. One block = 64 edges.
// Each wave owns 16 output cols; per K-step: 1 B-frag load feeds 4 WMMAs.
// ---------------------------------------------------------------------------
__global__ void __launch_bounds__(256) embed_kernel(
    const f16* __restrict__ xh, const int* __restrict__ src, const int* __restrict__ dst,
    const float* __restrict__ ea,
    const f16* __restrict__ w1P /*packed 64x128*/, const float* __restrict__ w1e,
    const float* __restrict__ b1,
    const float* __restrict__ lng, const float* __restrict__ lnb,
    const f16* __restrict__ w2P /*packed 128x128*/, const float* __restrict__ b2,
    float* __restrict__ hout, int E) {
  __shared__ __attribute__((aligned(32))) f16   aP[64 * 64];     // frag-major A (GEMM1)
  __shared__ float msg[64 * 128];
  __shared__ __attribute__((aligned(32))) f16   msghP[64 * 128]; // frag-major A (GEMM2)
  __shared__ float red[640];
  __shared__ int   sidx[64];
  __shared__ int   didx[64];
  __shared__ float eav[64];

  int tid = threadIdx.x;
  int e0  = blockIdx.x * 64;
  if (tid < 64) {
    int e = e0 + tid;
    bool ok = e < E;
    sidx[tid] = ok ? src[e] : 0;
    didx[tid] = ok ? dst[e] : -1;
    eav[tid]  = ok ? ea[e] : 0.f;
  }
  __syncthreads();
  // Gather x[src] rows into fragment-major LDS.
  for (int t = tid; t < 64 * 64; t += 256) {
    int r = t >> 6, c = t & 63;
    f16 v = (e0 + r < E) ? xh[(size_t)sidx[r] * 64 + c] : (f16)0.f;
    int rt = r >> 4, m = r & 15, kt = c >> 5, kk = c & 31;
    aP[(rt * 2 + kt) * 512 + afrag_idx(m, kk)] = v;
  }
  __syncthreads();

  int wave = tid >> 5, lane = tid & 31;
  int n  = wave * 16 + (lane & 15);
  int mh = (lane >> 4) * 8;

  // GEMM1: [64x64] x [64x128] + b1 + ea*w1[64,:], relu
  {
    v8f acc[4] = {};
#pragma unroll
    for (int kt = 0; kt < 2; ++kt) {
      __builtin_prefetch(w2P + ((size_t)(kt * 8 + wave)) * 512, 0, 3);  // warm GEMM2 B
      v16h bf = load_b_packed(w1P, 8, kt, wave);
#pragma unroll
      for (int rt = 0; rt < 4; ++rt)
        acc[rt] = wmma_acc(load_a_packed(aP, 2, rt, kt), bf, acc[rt]);
    }
#pragma unroll
    for (int rt = 0; rt < 4; ++rt)
#pragma unroll
      for (int i = 0; i < 8; ++i) {
        int m = rt * 16 + mh + i;
        float v = acc[rt][i] + b1[n] + eav[m] * w1e[n];
        msg[m * 128 + n] = fmaxf(v, 0.f);
      }
  }
  __syncthreads();
  ln_rows64x128(msg, lng, lnb, red, msghP);

  // GEMM2: [64x128] x [128x128] + b2, relu
  {
    v8f acc[4] = {};
#pragma unroll
    for (int kt = 0; kt < 4; ++kt) {
      v16h bf = load_b_packed(w2P, 8, kt, wave);
#pragma unroll
      for (int rt = 0; rt < 4; ++rt)
        acc[rt] = wmma_acc(load_a_packed(msghP, 4, rt, kt), bf, acc[rt]);
    }
#pragma unroll
    for (int rt = 0; rt < 4; ++rt)
#pragma unroll
      for (int i = 0; i < 8; ++i) {
        int m = rt * 16 + mh + i;
        msg[m * 128 + n] = fmaxf(acc[rt][i] + b2[n], 0.f);
      }
  }
  __syncthreads();
  ln_rows64x128(msg, lng, lnb, red, nullptr);

  // Scatter-add 64x128 messages to destination nodes.
  for (int t = tid; t < 64 * 128; t += 256) {
    int r = t >> 7, c = t & 127;
    int d = didx[r];
    if (d >= 0) atomicAdd(&hout[(size_t)d * 128 + c], msg[t]);
  }
}

// ---------------------------------------------------------------------------
// Node projections: Y = X @ W + b for 4 weights (q,k,v,skip). One block = 64 nodes.
// ---------------------------------------------------------------------------
__global__ void __launch_bounds__(256) node_gemm4_kernel(
    const f16* __restrict__ hh,
    const f16* wA, const f16* wB, const f16* wC, const f16* wD,
    const float* bA, const float* bB, const float* bC, const float* bD,
    float* oA, float* oB, float* oC, float* oD, int n_nodes) {
  __shared__ __attribute__((aligned(32))) f16 aP[64 * 128];
  const f16* w; const float* bias; float* out;
  switch (blockIdx.y) {
    case 0:  w = wA; bias = bA; out = oA; break;
    case 1:  w = wB; bias = bB; out = oB; break;
    case 2:  w = wC; bias = bC; out = oC; break;
    default: w = wD; bias = bD; out = oD; break;
  }
  int tid = threadIdx.x;
  int r0  = blockIdx.x * 64;
  for (int t = tid; t < 64 * 128; t += 256) {
    int r = t >> 7, c = t & 127;
    f16 v = (r0 + r < n_nodes) ? hh[(size_t)(r0 + r) * 128 + c] : (f16)0.f;
    int rt = r >> 4, m = r & 15, kt = c >> 5, kk = c & 31;
    aP[(rt * 4 + kt) * 512 + afrag_idx(m, kk)] = v;
  }
  __syncthreads();
  int wave = tid >> 5, lane = tid & 31;
  v8f acc[4] = {};
#pragma unroll
  for (int kt = 0; kt < 4; ++kt) {
    if (kt < 3) __builtin_prefetch(w + ((size_t)((kt + 1) * 8 + wave)) * 512, 0, 3);
    v16h bf = load_b_packed(w, 8, kt, wave);
#pragma unroll
    for (int rt = 0; rt < 4; ++rt)
      acc[rt] = wmma_acc(load_a_packed(aP, 4, rt, kt), bf, acc[rt]);
  }
  int n  = wave * 16 + (lane & 15);
  int mh = (lane >> 4) * 8;
#pragma unroll
  for (int rt = 0; rt < 4; ++rt)
#pragma unroll
    for (int i = 0; i < 8; ++i) {
      int r = r0 + rt * 16 + mh + i;
      if (r < n_nodes) out[(size_t)r * 128 + n] = acc[rt][i] + bias[n];
    }
}

// ---------------------------------------------------------------------------
// Attention: logits + segment max, exp + segment sum, weighted scatter.
// ---------------------------------------------------------------------------
__global__ void __launch_bounds__(256) edge_logit_kernel(
    const float* __restrict__ q, const float* __restrict__ k,
    const int* __restrict__ src, const int* __restrict__ dst,
    const float* __restrict__ ea, const float* __restrict__ we,
    float* __restrict__ logit, unsigned* __restrict__ mkey, int E) {
  long gid = (long)blockIdx.x * 256 + threadIdx.x;
  if (gid >= (long)E * 4) return;
  int e = (int)(gid >> 2), h = (int)(gid & 3);
  int s = src[e], d = dst[e];
  float av = ea[e];
  const float* qp = q + (size_t)d * 128 + h * 32;
  const float* kp = k + (size_t)s * 128 + h * 32;
  const float* wp = we + h * 32;
  float acc = 0.f;
#pragma unroll
  for (int c = 0; c < 32; ++c) acc += qp[c] * (kp[c] + av * wp[c]);
  acc *= 0.17677669529663687f;  // 1/sqrt(32)
  logit[gid] = acc;
  atomicMax(&mkey[(size_t)d * 4 + h], fkey(acc));
}

__global__ void __launch_bounds__(256) edge_exp_kernel(
    const float* __restrict__ logit, const int* __restrict__ dst,
    const unsigned* __restrict__ mkey, float* __restrict__ aval,
    float* __restrict__ ssum, int E) {
  long gid = (long)blockIdx.x * 256 + threadIdx.x;
  if (gid >= (long)E * 4) return;
  int e = (int)(gid >> 2), h = (int)(gid & 3);
  int d = dst[e];
  float m = funkey(mkey[(size_t)d * 4 + h]);
  float a = __expf(logit[gid] - m);
  aval[gid] = a;
  atomicAdd(&ssum[(size_t)d * 4 + h], a);
}

__global__ void __launch_bounds__(256) edge_agg_kernel(
    const float* __restrict__ v, const int* __restrict__ src, const int* __restrict__ dst,
    const float* __restrict__ ea, const float* __restrict__ we,
    const float* __restrict__ aval, const float* __restrict__ ssum,
    float* __restrict__ agg, int E) {
  long gid = (long)blockIdx.x * 256 + threadIdx.x;
  if (gid >= (long)E * 128) return;
  int e = (int)(gid >> 7), c = (int)(gid & 127), h = c >> 5;
  int s = src[e], d = dst[e];
  float alpha = aval[(size_t)e * 4 + h] / (ssum[(size_t)d * 4 + h] + 1e-16f);
  float ve = v[(size_t)s * 128 + c] + ea[e] * we[c];
  atomicAdd(&agg[(size_t)d * 128 + c], alpha * ve);
}

// ---------------------------------------------------------------------------
// Utility kernels
// ---------------------------------------------------------------------------
__global__ void __launch_bounds__(256) zero_f32_kernel(float* p, long n) {
  long i = (long)blockIdx.x * 256 + threadIdx.x;
  if (i < n) p[i] = 0.f;
}
__global__ void __launch_bounds__(256) f32_to_f16_kernel(const float* in, f16* out, long n) {
  long i = (long)blockIdx.x * 256 + threadIdx.x;
  if (i < n) out[i] = (f16)in[i];
}
__global__ void __launch_bounds__(256) relu_cvt_kernel(float* buf, f16* outh, long n) {
  long i = (long)blockIdx.x * 256 + threadIdx.x;
  if (i < n) {
    float v = fmaxf(buf[i], 0.f);
    buf[i] = v;
    outh[i] = (f16)v;
  }
}
__global__ void __launch_bounds__(256) pool_kernel(const float* __restrict__ h,
                                                   const int* __restrict__ batch,
                                                   float* __restrict__ pool,
                                                   int* __restrict__ cnt, int n) {
  long gid = (long)blockIdx.x * 256 + threadIdx.x;
  if (gid >= (long)n * 128) return;
  int node = (int)(gid >> 7), c = (int)(gid & 127);
  int b = batch[node];
  atomicAdd(&pool[b * 128 + c], h[(size_t)node * 128 + c]);
  if (c == 0) atomicAdd(&cnt[b], 1);
}
__global__ void __launch_bounds__(256) pool_div_kernel(const float* pool, const int* cnt,
                                                       float* out) {
  int gid = blockIdx.x * 256 + threadIdx.x;
  if (gid >= 8 * 128) return;
  out[gid] = pool[gid] / fmaxf((float)cnt[gid >> 7], 1.f);
}

// ---------------------------------------------------------------------------
// Launch
// ---------------------------------------------------------------------------
extern "C" void kernel_launch(void* const* d_in, const int* in_sizes, int n_in,
                              void* d_out, int out_size, void* d_ws, size_t ws_size,
                              hipStream_t stream) {
  const float* x     = (const float*)d_in[0];
  const int*   eidx  = (const int*)d_in[1];
  const float* ea    = (const float*)d_in[2];
  const int*   batch = (const int*)d_in[3];
  const float* w1    = (const float*)d_in[4];
  const float* b1    = (const float*)d_in[5];
  const float* lng   = (const float*)d_in[6];
  const float* lnb   = (const float*)d_in[7];
  const float* w2    = (const float*)d_in[8];
  const float* b2    = (const float*)d_in[9];

  int N = in_sizes[0] / 64;
  int E = in_sizes[2];
  const int* src = eidx;
  const int* dst = eidx + E;

  char* wsb = (char*)d_ws;
  size_t off = 0;
  auto alloc = [&](size_t bytes) -> void* {
    void* p = wsb + off;
    off = (off + bytes + 255) & ~(size_t)255;
    return p;
  };
  f16* xh  = (f16*)alloc((size_t)N * 64 * 2);
  f16* hh  = (f16*)alloc((size_t)N * 128 * 2);
  f16* w1P = (f16*)alloc(64 * 128 * 2);
  f16* w2P = (f16*)alloc(128 * 128 * 2);
  f16* wt[8];
  for (int i = 0; i < 8; ++i) wt[i] = (f16*)alloc(128 * 128 * 2);
  float*    bufH  = (float*)alloc((size_t)N * 128 * 4);  // embed out / layer2 agg
  float*    bufA  = (float*)alloc((size_t)N * 128 * 4);  // layer1 agg / layer2 input
  float*    qb    = (float*)alloc((size_t)N * 128 * 4);
  float*    kb    = (float*)alloc((size_t)N * 128 * 4);
  float*    vb    = (float*)alloc((size_t)N * 128 * 4);
  float*    logit = (float*)alloc((size_t)E * 4 * 4);
  float*    aval  = (float*)alloc((size_t)E * 4 * 4);
  unsigned* mkey  = (unsigned*)alloc((size_t)N * 4 * 4);
  float*    ssum  = (float*)alloc((size_t)N * 4 * 4);
  float*    pool  = (float*)alloc(8 * 128 * 4);
  int*      cnt   = (int*)alloc(8 * 4);

  dim3 B(256);
  long nh = (long)N * 128;
  auto g = [](long n) { return dim3((unsigned)((n + 255) / 256)); };

  // Zero accumulators.
  zero_f32_kernel<<<g(nh), B, 0, stream>>>(bufH, nh);
  zero_f32_kernel<<<g((long)N * 4), B, 0, stream>>>((float*)mkey, (long)N * 4);
  zero_f32_kernel<<<g((long)N * 4), B, 0, stream>>>(ssum, (long)N * 4);
  zero_f32_kernel<<<g(1024), B, 0, stream>>>(pool, 1024);
  zero_f32_kernel<<<g(8), B, 0, stream>>>((float*)cnt, 8);

  // f16 conversion of x; fragment-major repack of all GEMM weights.
  f32_to_f16_kernel<<<g((long)N * 64), B, 0, stream>>>(x, xh, (long)N * 64);
  pack_b_kernel<<<g(64 * 128), B, 0, stream>>>(w1, w1P, 64, 128);  // rows 0..63 of W1
  pack_b_kernel<<<g(128 * 128), B, 0, stream>>>(w2, w2P, 128, 128);
  const float* wsrc[8] = {(const float*)d_in[10], (const float*)d_in[12],
                          (const float*)d_in[14], (const float*)d_in[17],
                          (const float*)d_in[19], (const float*)d_in[21],
                          (const float*)d_in[23], (const float*)d_in[26]};
  for (int i = 0; i < 8; ++i)
    pack_b_kernel<<<g(128 * 128), B, 0, stream>>>(wsrc[i], wt[i], 128, 128);

  // Embed conv: ~40 GFLOP of WMMA + scatter-add.
  const float* w1e = w1 + 64 * 128;  // edge-attr row of W1
  embed_kernel<<<dim3((E + 63) / 64), B, 0, stream>>>(xh, src, dst, ea, w1P, w1e, b1,
                                                      lng, lnb, w2P, b2, bufH, E);
  f32_to_f16_kernel<<<g(nh), B, 0, stream>>>(bufH, hh, nh);

  // Two TransformerConv layers.
  for (int layer = 0; layer < 2; ++layer) {
    const float* bq = (const float*)d_in[layer ? 20 : 11];
    const float* bk = (const float*)d_in[layer ? 22 : 13];
    const float* bv = (const float*)d_in[layer ? 24 : 15];
    const float* we = (const float*)d_in[layer ? 25 : 16];
    const float* bs = (const float*)d_in[layer ? 27 : 18];
    f16* wq = wt[layer * 4 + 0];
    f16* wk = wt[layer * 4 + 1];
    f16* wv = wt[layer * 4 + 2];
    f16* wsk = wt[layer * 4 + 3];
    float* agg = layer ? bufH : bufA;  // skip written here; scatter adds on top

    node_gemm4_kernel<<<dim3((N + 63) / 64, 4), B, 0, stream>>>(
        hh, wq, wk, wv, wsk, bq, bk, bv, bs, qb, kb, vb, agg, N);

    edge_logit_kernel<<<g((long)E * 4), B, 0, stream>>>(qb, kb, src, dst, ea, we, logit,
                                                        mkey, E);
    edge_exp_kernel<<<g((long)E * 4), B, 0, stream>>>(logit, dst, mkey, aval, ssum, E);
    edge_agg_kernel<<<g((long)E * 128), B, 0, stream>>>(vb, src, dst, ea, we, aval, ssum,
                                                        agg, E);
    relu_cvt_kernel<<<g(nh), B, 0, stream>>>(agg, hh, nh);

    if (layer == 0) {  // reset segment max/sum for layer 2
      zero_f32_kernel<<<g((long)N * 4), B, 0, stream>>>((float*)mkey, (long)N * 4);
      zero_f32_kernel<<<g((long)N * 4), B, 0, stream>>>(ssum, (long)N * 4);
    }
  }

  // Global mean pool over batch_idx -> [8,128].
  pool_kernel<<<g(nh), B, 0, stream>>>(bufH, batch, pool, cnt, N);
  pool_div_kernel<<<g(1024), B, 0, stream>>>(pool, cnt, (float*)d_out);
}
